// MultiHeadAttention_26010321944593
// MI455X (gfx1250) — matrix operand order repaired and checked by reading, not compiled, not run on hardware
//
#include <hip/hip_runtime.h>
#include <hip/hip_bf16.h>

// -------------------------------------------------------------------------
// MultiHeadAttention forward for MI455X (gfx1250, wave32, WMMA bf16).
// Stages: cvt(fp32->bf16) -> QKV gemm -> flash attention -> proj gemm.
// All matrix math on v_wmma_f32_16x16x32_bf16 with fp32 accumulation.
// Attention K/V tiles are staged to LDS with global_load_async_to_lds_b128
// (ASYNCcnt + s_wait_asynccnt), double-buffered across the key loop.
// Workspace layout (needs ~48MB):
//   [0,8M)    x_bf16          [4096,1024]
//   [8M,14M)  w_qkv_bf16      [3072,1024]
//   [14M,16M) w_proj_bf16     [1024,1024]
//   [16M,24M) Q  bf16         [H][N][64]  (pre-scaled by 1/sqrt(D))
//   [24M,32M) K  bf16         [H][N][64]
//   [32M,40M) V^T bf16        [H][64][N]
//   [40M,48M) O  bf16         [4096,1024]
// -------------------------------------------------------------------------

typedef __bf16 bhalf;
typedef bhalf bf16x4  __attribute__((ext_vector_type(4)));
typedef bhalf bf16x8  __attribute__((ext_vector_type(8)));
typedef bhalf bf16x16 __attribute__((ext_vector_type(16)));
typedef float v8f     __attribute__((ext_vector_type(8)));

#define N_SEQ 4096
#define EMB   1024
#define NH    16
#define HD    64
#define F3    3072
#define KSTR  72   // LDS row pitch (elements): 36 dwords -> conflict-free frags

static __device__ inline v8f wmma_bf16(bf16x16 a, bf16x16 b, v8f c) {
  // D = A(16x32) * B(32x16) + C, fp32 accumulate
  return __builtin_amdgcn_wmma_f32_16x16x32_bf16(
      false, a, false, b, (short)0, c, false, false);
}

// Async global->LDS copy, 16B per lane. VDST = LDS byte address (low 32 bits
// of the generic pointer ARE the LDS address per ISA flat addressing rules).
static __device__ inline void async_b128(unsigned lds_addr, const void* gptr) {
  asm volatile("global_load_async_to_lds_b128 %0, %1, off"
               :: "v"(lds_addr), "v"(gptr) : "memory");
}

// A-matrix fragment (16x32 bf16, row-major source, lda in elements):
// lane L holds row m0+(L&15); elements 0..7 = K kb..kb+7, 8..15 = kb+16..kb+23
// with kb = k0 + (L>=16 ? 8 : 0).
static __device__ inline bf16x16 load_a_frag(const bhalf* __restrict__ A,
                                             int lda, int m0, int k0, int lane) {
  int row = m0 + (lane & 15);
  int kb  = k0 + ((lane >> 4) << 3);
  const bhalf* p = A + row * lda + kb;
  bf16x8 lo = *(const bf16x8*)(p);
  bf16x8 hi = *(const bf16x8*)(p + 16);
  return __builtin_shufflevector(lo, hi, 0,1,2,3,4,5,6,7,8,9,10,11,12,13,14,15);
}

// B-matrix fragment (32x16 bf16). B[k][n] = W[n0+n][k0+k] with W row-major
// (ldw elements): lane L holds column n0+(L&15); elements 0..7 = K kb..kb+7,
// 8..15 = kb+16..kb+23 with kb = k0 + (L>=16 ? 8 : 0).
static __device__ inline bf16x16 load_b_frag(const bhalf* __restrict__ W,
                                             int ldw, int n0, int k0, int lane) {
  int row = n0 + (lane & 15);
  int kb  = k0 + ((lane >> 4) << 3);
  const bhalf* p = W + row * ldw + kb;
  bf16x8 lo = *(const bf16x8*)(p);
  bf16x8 hi = *(const bf16x8*)(p + 16);
  return __builtin_shufflevector(lo, hi, 0,1,2,3,4,5,6,7,8,9,10,11,12,13,14,15);
}

// Same fragment shapes sourced from LDS tiles with KSTR row pitch.
static __device__ inline bf16x16 lds_a_frag(const bhalf* base, int row, int kb) {
  const bhalf* p = base + row * KSTR + kb;
  bf16x8 lo = *(const bf16x8*)(p);
  bf16x8 hi = *(const bf16x8*)(p + 16);
  return __builtin_shufflevector(lo, hi, 0,1,2,3,4,5,6,7,8,9,10,11,12,13,14,15);
}
static __device__ inline bf16x16 lds_b_frag(const bhalf* base, int row, int kb) {
  const bhalf* p = base + row * KSTR + kb;
  bf16x8 lo = *(const bf16x8*)(p);
  bf16x8 hi = *(const bf16x8*)(p + 16);
  return __builtin_shufflevector(lo, hi, 0,1,2,3,4,5,6,7,8,9,10,11,12,13,14,15);
}

// ---------------- fp32 -> bf16 convert (4 elems / thread) -----------------
__global__ void cvt_bf16_kernel(const float* __restrict__ src,
                                bhalf* __restrict__ dst, int n4) {
  int i = blockIdx.x * blockDim.x + threadIdx.x;
  if (i < n4) {
    float4 v = ((const float4*)src)[i];
    bf16x4 o;
    o[0] = (bhalf)v.x; o[1] = (bhalf)v.y; o[2] = (bhalf)v.z; o[3] = (bhalf)v.w;
    ((bf16x4*)dst)[i] = o;
  }
}

// ---------------- QKV gemm: qkv[n,f] = sum_e x[n,e]*w_qkv[f,e] + b ---------
// Block 256 thr = 8 waves (4 M x 2 N), block tile 128x128, wave tile 32x64.
__global__ __launch_bounds__(256)
void qkv_gemm_kernel(const bhalf* __restrict__ X, const bhalf* __restrict__ Wq,
                     const float* __restrict__ bias,
                     bhalf* __restrict__ Qh, bhalf* __restrict__ Kh,
                     bhalf* __restrict__ Vt) {
  const int lane = threadIdx.x & 31;
  const int wid  = threadIdx.x >> 5;
  const int m0   = blockIdx.y * 128 + (wid >> 1) * 32;
  const int f0   = blockIdx.x * 128 + (wid & 1) * 64;

  v8f acc[2][4];
#pragma unroll
  for (int mt = 0; mt < 2; ++mt)
#pragma unroll
    for (int nt = 0; nt < 4; ++nt)
      acc[mt][nt] = (v8f){0.f,0.f,0.f,0.f,0.f,0.f,0.f,0.f};

  for (int k0 = 0; k0 < EMB; k0 += 32) {
    bf16x16 a0 = load_a_frag(X, EMB, m0,      k0, lane);
    bf16x16 a1 = load_a_frag(X, EMB, m0 + 16, k0, lane);
#pragma unroll
    for (int nt = 0; nt < 4; ++nt) {
      bf16x16 b = load_b_frag(Wq, EMB, f0 + nt * 16, k0, lane);
      acc[0][nt] = wmma_bf16(a0, b, acc[0][nt]);
      acc[1][nt] = wmma_bf16(a1, b, acc[1][nt]);
    }
  }

  const float qscale = 0.125f;  // 1/sqrt(64)
#pragma unroll
  for (int mt = 0; mt < 2; ++mt) {
#pragma unroll
    for (int nt = 0; nt < 4; ++nt) {
      int f     = f0 + nt * 16 + (lane & 15);
      float bv  = bias[f];
      int which = f >> 10;       // 0:Q 1:K 2:V
      int fi    = f & 1023;
      int h     = fi >> 6;
      int d     = fi & 63;
#pragma unroll
      for (int r = 0; r < 8; ++r) {
        int row = m0 + mt * 16 + ((lane >> 4) << 3) + r;
        float v = acc[mt][nt][r] + bv;
        if (which == 0)
          Qh[((h * N_SEQ + row) << 6) + d] = (bhalf)(v * qscale);
        else if (which == 1)
          Kh[((h * N_SEQ + row) << 6) + d] = (bhalf)v;
        else
          Vt[((h << 6) + d) * N_SEQ + row] = (bhalf)v;  // store V transposed
      }
    }
  }
}

// ---------------- Flash attention ------------------------------------------
// Grid (N/64, H), block = 128 = 4 waves; each wave owns 16 query rows.
// S^T = K * Q^T  (M=keys, N=queries) so C-layout of S^T == A-layout of P
// for O = P * V (no LDS transpose). Softmax stats are per-lane scalars.
// K/V tiles staged to LDS with async loads, double-buffered.
__global__ __launch_bounds__(128)
void attn_kernel(const bhalf* __restrict__ Qh, const bhalf* __restrict__ Kh,
                 const bhalf* __restrict__ Vt, bhalf* __restrict__ Ob) {
  __shared__ bhalf Kbuf[2][64 * KSTR];   // [buf][key_local][d]
  __shared__ bhalf Vbuf[2][64 * KSTR];   // [buf][d][key_local]

  const int tid  = threadIdx.x;
  const int lane = tid & 31;
  const int wid  = tid >> 5;
  const int h    = blockIdx.y;
  const int q0   = blockIdx.x * 64 + wid * 16;

  const bhalf* Qp = Qh + h * N_SEQ * HD;
  const bhalf* Kp = Kh + h * N_SEQ * HD;
  const bhalf* Vp = Vt + h * HD * N_SEQ;   // [64][N]

  // Stage one 64-key tile (K: 64x64 contiguous; V^T: 64 rows, stride N).
  // 512 x 16B chunks per matrix, 128 threads -> 4 chunks each -> 8 async
  // instructions per wave per tile.
  auto stage = [&](int j, int buf) {
    bhalf* Kl = &Kbuf[buf][0];
    bhalf* Vl = &Vbuf[buf][0];
    const bhalf* Kg = Kp + j * HD;
#pragma unroll
    for (int i = 0; i < 4; ++i) {
      int c    = i * 128 + tid;     // 0..511
      int row  = c >> 3;
      int part = (c & 7) * 8;       // element offset within row
      async_b128((unsigned)(uintptr_t)(Kl + row * KSTR + part),
                 (const void*)(Kg + row * HD + part));
      async_b128((unsigned)(uintptr_t)(Vl + row * KSTR + part),
                 (const void*)(Vp + row * N_SEQ + j + part));
    }
  };

  // Q^T as B-matrix, fixed for the whole loop (two 32-deep k-steps over D=64)
  bf16x16 qb[2];
#pragma unroll
  for (int s = 0; s < 2; ++s) qb[s] = load_b_frag(Qp, HD, q0, s * 32, lane);

  v8f o[4];
#pragma unroll
  for (int t = 0; t < 4; ++t) o[t] = (v8f){0.f,0.f,0.f,0.f,0.f,0.f,0.f,0.f};
  float m = -1e30f, l = 0.f;
  const int hb = (lane >> 4) << 3;

  stage(0, 0);  // prefetch first tile

  for (int j = 0; j < N_SEQ; j += 64) {
    const int buf = (j >> 6) & 1;
    if (j + 64 < N_SEQ) {
      stage(j + 64, buf ^ 1);                            // prefetch next tile
      asm volatile("s_wait_asynccnt 0x8" ::: "memory");  // current tile landed
    } else {
      asm volatile("s_wait_asynccnt 0x0" ::: "memory");
    }
    __syncthreads();   // all waves' async data visible

    const bhalf* Kl = &Kbuf[buf][0];
    const bhalf* Vl = &Vbuf[buf][0];

    // S^T block: 64 keys x 16 queries, 4 M-tiles
    v8f s4[4];
#pragma unroll
    for (int t = 0; t < 4; ++t) s4[t] = (v8f){0.f,0.f,0.f,0.f,0.f,0.f,0.f,0.f};
#pragma unroll
    for (int t = 0; t < 4; ++t)
#pragma unroll
      for (int st = 0; st < 2; ++st) {
        bf16x16 kf = lds_a_frag(Kl, t * 16 + (lane & 15),
                                st * 32 + ((lane >> 4) << 3));
        s4[t] = wmma_bf16(kf, qb[st], s4[t]);
      }

    // online softmax; lane owns query column q0+(lane&15)
    float bm = -1e30f;
#pragma unroll
    for (int t = 0; t < 4; ++t)
#pragma unroll
      for (int r = 0; r < 8; ++r) bm = fmaxf(bm, s4[t][r]);
    bm = fmaxf(bm, __shfl_xor(bm, 16, 32));
    float mn   = fmaxf(m, bm);
    float corr = __expf(m - mn);
    m = mn;

    float rs = 0.f;
    bf16x16 pa[2];   // P as A-matrix fragments for O += P*V
#pragma unroll
    for (int s = 0; s < 2; ++s)
#pragma unroll
      for (int r = 0; r < 8; ++r) {
        float p0 = __expf(s4[2 * s][r] - m);
        float p1 = __expf(s4[2 * s + 1][r] - m);
        rs += p0 + p1;
        pa[s][r]     = (bhalf)p0;
        pa[s][r + 8] = (bhalf)p1;
      }
    rs += __shfl_xor(rs, 16, 32);
    l = l * corr + rs;

    // broadcast per-row correction to the O-accumulator layout
    float corrv[8];
#pragma unroll
    for (int r = 0; r < 8; ++r) corrv[r] = __shfl(corr, hb + r, 32);

#pragma unroll
    for (int t = 0; t < 4; ++t) {
#pragma unroll
      for (int r = 0; r < 8; ++r) o[t][r] *= corrv[r];
#pragma unroll
      for (int s = 0; s < 2; ++s) {
        bf16x16 vf = lds_b_frag(Vl, t * 16 + (lane & 15),
                                s * 32 + ((lane >> 4) << 3));
        o[t] = wmma_bf16(pa[s], vf, o[t]);
      }
    }
    __syncthreads();   // done reading this buffer before it is overwritten
  }

  float linv = 1.0f / l;
  float lv[8];
#pragma unroll
  for (int r = 0; r < 8; ++r) lv[r] = __shfl(linv, hb + r, 32);
#pragma unroll
  for (int t = 0; t < 4; ++t)
#pragma unroll
    for (int r = 0; r < 8; ++r) {
      int row = q0 + hb + r;
      int col = h * HD + t * 16 + (lane & 15);
      Ob[row * EMB + col] = (bhalf)(o[t][r] * lv[r]);
    }
}

// ---------------- out projection: out[n,f] = sum_e O[n,e]*w_proj[f,e]+b ----
__global__ __launch_bounds__(256)
void proj_gemm_kernel(const bhalf* __restrict__ Ob, const bhalf* __restrict__ Wp,
                      const float* __restrict__ bias, float* __restrict__ out) {
  const int lane = threadIdx.x & 31;
  const int wid  = threadIdx.x >> 5;
  const int m0   = blockIdx.y * 128 + (wid >> 1) * 32;
  const int f0   = blockIdx.x * 128 + (wid & 1) * 64;

  v8f acc[2][4];
#pragma unroll
  for (int mt = 0; mt < 2; ++mt)
#pragma unroll
    for (int nt = 0; nt < 4; ++nt)
      acc[mt][nt] = (v8f){0.f,0.f,0.f,0.f,0.f,0.f,0.f,0.f};

  for (int k0 = 0; k0 < EMB; k0 += 32) {
    bf16x16 a0 = load_a_frag(Ob, EMB, m0,      k0, lane);
    bf16x16 a1 = load_a_frag(Ob, EMB, m0 + 16, k0, lane);
#pragma unroll
    for (int nt = 0; nt < 4; ++nt) {
      bf16x16 b = load_b_frag(Wp, EMB, f0 + nt * 16, k0, lane);
      acc[0][nt] = wmma_bf16(a0, b, acc[0][nt]);
      acc[1][nt] = wmma_bf16(a1, b, acc[1][nt]);
    }
  }

#pragma unroll
  for (int mt = 0; mt < 2; ++mt)
#pragma unroll
    for (int nt = 0; nt < 4; ++nt) {
      int f    = f0 + nt * 16 + (lane & 15);
      float bv = bias[f];
#pragma unroll
      for (int r = 0; r < 8; ++r) {
        int row = m0 + mt * 16 + ((lane >> 4) << 3) + r;
        out[row * EMB + f] = acc[mt][nt][r] + bv;
      }
    }
}

// ---------------------------------------------------------------------------
extern "C" void kernel_launch(void* const* d_in, const int* in_sizes, int n_in,
                              void* d_out, int out_size, void* d_ws, size_t ws_size,
                              hipStream_t stream) {
  const float* x      = (const float*)d_in[0];
  const float* w_qkv  = (const float*)d_in[1];
  const float* b_qkv  = (const float*)d_in[2];
  const float* w_proj = (const float*)d_in[3];
  const float* b_proj = (const float*)d_in[4];
  float* out = (float*)d_out;

  char* ws = (char*)d_ws;
  const size_t MB = 1u << 20;
  bhalf* xb     = (bhalf*)(ws + 0 * MB);
  bhalf* wqkvb  = (bhalf*)(ws + 8 * MB);
  bhalf* wprojb = (bhalf*)(ws + 14 * MB);
  bhalf* Qh     = (bhalf*)(ws + 16 * MB);
  bhalf* Kh     = (bhalf*)(ws + 24 * MB);
  bhalf* Vt     = (bhalf*)(ws + 32 * MB);
  bhalf* Ob     = (bhalf*)(ws + 40 * MB);

  // fp32 -> bf16 conversions
  {
    int n4 = (N_SEQ * EMB) / 4;
    cvt_bf16_kernel<<<(n4 + 255) / 256, 256, 0, stream>>>(x, xb, n4);
  }
  {
    int n4 = (F3 * EMB) / 4;
    cvt_bf16_kernel<<<(n4 + 255) / 256, 256, 0, stream>>>(w_qkv, wqkvb, n4);
  }
  {
    int n4 = (EMB * EMB) / 4;
    cvt_bf16_kernel<<<(n4 + 255) / 256, 256, 0, stream>>>(w_proj, wprojb, n4);
  }

  // QKV projection (writes head-major Q, K and transposed V)
  qkv_gemm_kernel<<<dim3(F3 / 128, N_SEQ / 128), 256, 0, stream>>>(
      xb, wqkvb, b_qkv, Qh, Kh, Vt);

  // Flash attention (async-LDS staged K/V, double-buffered)
  attn_kernel<<<dim3(N_SEQ / 64, NH), 128, 0, stream>>>(Qh, Kh, Vt, Ob);

  // Output projection (fp32 out + bias)
  proj_gemm_kernel<<<dim3(EMB / 128, N_SEQ / 128), 256, 0, stream>>>(
      Ob, wprojb, b_proj, out);
}
